// UniGCNII_67688684585240
// MI455X (gfx1250) — matrix operand order repaired and checked by reading, not compiled, not run on hardware
//
#include <hip/hip_runtime.h>
#include <cstdint>

typedef __attribute__((ext_vector_type(16))) _Float16 v16h;
typedef __attribute__((ext_vector_type(8)))  _Float16 v8h;
typedef __attribute__((ext_vector_type(8)))  float    v8f;

#define DFEAT 256  // NFEAT == NHID == 256

__device__ __forceinline__ float silu_f(float x) { return x / (1.0f + __expf(-x)); }

// ------------------------------------------------------------------
// WMMA GEMM:  out = epilogue( A[nrows,256] @ W[validCols,256]^T )
// MODE 0: out = silu(acc + bias[n])  -> writes out (X) and out2 (X0), stride 256
// MODE 1: out = silu((1-beta)*Xi + beta*acc) -> writes out (X), stride 256
// MODE 2: out[m*64+n] = acc + bias[n]  for n < validCols (logits, stride 64)
// Block: 256 threads (8 waves), 64 rows x NCOLS cols. Waves tiled 2(M) x 4(N).
// Entire W matrix converted fp32->f16 into LDS once; k-loop is barrier-free.
// ------------------------------------------------------------------
template<int NCOLS, int MODE>
__global__ __launch_bounds__(256)
void gemm_kernel(const float* __restrict__ A,
                 const float* __restrict__ Wg,
                 int validCols,
                 const float* __restrict__ bias,
                 const float* __restrict__ Xi,
                 const float* __restrict__ beta_arr, int layer,
                 float* __restrict__ out,
                 float* __restrict__ out2,
                 int nrows)
{
    constexpr int TN  = NCOLS / 64;   // 16-col tiles per wave
    constexpr int WST = 264;          // halves per W row in LDS (256 + 8 pad)
    __shared__ _Float16 Wlds[NCOLS * WST];

    const int lane  = threadIdx.x & 31;
    const int wave  = threadIdx.x >> 5;
    const int waveM = wave >> 2;      // 0..1
    const int waveN = wave & 3;       // 0..3
    const int rowBlk = blockIdx.x * 64;

    float beta = 0.0f;
    if constexpr (MODE == 1) beta = beta_arr[layer];

    // ---- one-time cooperative preload: whole W, fp32 -> f16, row stride WST ----
    // Each iteration moves one aligned 8-half chunk: 2x global b128 -> 1x ds b128.
    #pragma unroll 4
    for (int idx = threadIdx.x; idx < NCOLS * 32; idx += 256) {
        int c  = idx >> 5;          // W row (output column)
        int kc = (idx & 31) << 3;   // k base within row, multiple of 8
        float4 w0 = make_float4(0.f, 0.f, 0.f, 0.f);
        float4 w1 = make_float4(0.f, 0.f, 0.f, 0.f);
        bool valid = (MODE == 2) ? (c < validCols) : true;
        if (valid) {
            const float* p = Wg + (size_t)c * DFEAT + kc;
            w0 = *(const float4*)(p);
            w1 = *(const float4*)(p + 4);
        }
        v8h h;
        h[0] = (_Float16)w0.x; h[1] = (_Float16)w0.y;
        h[2] = (_Float16)w0.z; h[3] = (_Float16)w0.w;
        h[4] = (_Float16)w1.x; h[5] = (_Float16)w1.y;
        h[6] = (_Float16)w1.z; h[7] = (_Float16)w1.w;
        *(v8h*)&Wlds[c * WST + kc] = h;
    }
    __syncthreads();

    v8f zero = {};
    v8f acc[2][TN];
    #pragma unroll
    for (int i = 0; i < 2; ++i)
        #pragma unroll
        for (int j = 0; j < TN; ++j) acc[i][j] = zero;

    // A-fragment addressing (16-bit A 16x32 layout: lane%16 = M row,
    // lane/16 selects K-halves {0..7,16..23} vs {8..15,24..31})
    const int mrow0 = rowBlk + waveM * 32 + (lane & 15);
    const int mrow1 = mrow0 + 16;
    const int mm0 = mrow0 < nrows ? mrow0 : nrows - 1;
    const int mm1 = mrow1 < nrows ? mrow1 : nrows - 1;
    const int ksub = (lane >> 4) * 8;

    // B-fragment LDS bases (lane%16 = N col, lane/16 selects K-halves {0..7,16..23} vs {8..15,24..31})
    int bbase[TN];
    #pragma unroll
    for (int tn = 0; tn < TN; ++tn) {
        int ncol = waveN * (NCOLS / 4) + tn * 16 + (lane & 15);
        bbase[tn] = ncol * WST + ((lane >> 4) << 3);
    }

    const float* arow0 = A + (size_t)mm0 * DFEAT + ksub;
    const float* arow1 = A + (size_t)mm1 * DFEAT + ksub;

    #pragma unroll
    for (int k0 = 0; k0 < DFEAT; k0 += 32) {
        // A fragments: direct global fp32 load + convert
        v16h afrag[2];
        const float* rows2[2] = { arow0 + k0, arow1 + k0 };
        #pragma unroll
        for (int t = 0; t < 2; ++t) {
            const float* base = rows2[t];
            float4 f0 = *(const float4*)(base);
            float4 f1 = *(const float4*)(base + 4);
            float4 f2 = *(const float4*)(base + 16);
            float4 f3 = *(const float4*)(base + 20);
            v16h a;
            a[0]=(_Float16)f0.x; a[1]=(_Float16)f0.y; a[2]=(_Float16)f0.z; a[3]=(_Float16)f0.w;
            a[4]=(_Float16)f1.x; a[5]=(_Float16)f1.y; a[6]=(_Float16)f1.z; a[7]=(_Float16)f1.w;
            a[8]=(_Float16)f2.x; a[9]=(_Float16)f2.y; a[10]=(_Float16)f2.z; a[11]=(_Float16)f2.w;
            a[12]=(_Float16)f3.x; a[13]=(_Float16)f3.y; a[14]=(_Float16)f3.z; a[15]=(_Float16)f3.w;
            afrag[t] = a;
        }

        #pragma unroll
        for (int tn = 0; tn < TN; ++tn) {
            const _Float16* bp = &Wlds[bbase[tn] + k0];
            v8h lo = *(const v8h*)bp;
            v8h hi = *(const v8h*)(bp + 16);
            v16h b;
            #pragma unroll
            for (int j = 0; j < 8; ++j) { b[j] = lo[j]; b[j + 8] = hi[j]; }
            acc[0][tn] = __builtin_amdgcn_wmma_f32_16x16x32_f16(
                false, afrag[0], false, b, (short)0, acc[0][tn], false, false);
            acc[1][tn] = __builtin_amdgcn_wmma_f32_16x16x32_f16(
                false, afrag[1], false, b, (short)0, acc[1][tn], false, false);
        }
    }

    // Epilogue: C/D layout: VGPR r, lanes 0-15 -> M=r, lanes 16-31 -> M=8+r; N=lane%16
    #pragma unroll
    for (int tm = 0; tm < 2; ++tm) {
        #pragma unroll
        for (int tn = 0; tn < TN; ++tn) {
            int n  = waveN * (NCOLS / 4) + tn * 16 + (lane & 15);
            int mb = rowBlk + waveM * 32 + tm * 16 + ((lane >> 4) << 3);
            #pragma unroll
            for (int r = 0; r < 8; ++r) {
                int m = mb + r;
                if (m >= nrows) continue;
                float v = acc[tm][tn][r];
                if constexpr (MODE == 0) {
                    float y = silu_f(v + bias[n]);
                    out[(size_t)m * DFEAT + n]  = y;
                    out2[(size_t)m * DFEAT + n] = y;
                } else if constexpr (MODE == 1) {
                    float xiv = Xi[(size_t)m * DFEAT + n];
                    out[(size_t)m * DFEAT + n] = silu_f((1.0f - beta) * xiv + beta * v);
                } else {
                    if (n < validCols) out[(size_t)m * 64 + n] = v + bias[n];
                }
            }
        }
    }
}

// ------------------------------------------------------------------
__global__ void zero_kernel(float* p, long n)
{
    long i = (long)blockIdx.x * blockDim.x + threadIdx.x;
    long stride = (long)gridDim.x * blockDim.x;
    for (; i < n; i += stride) p[i] = 0.0f;
}

__global__ void edge_count_kernel(const int* __restrict__ edges, float* cnt, int nnz)
{
    int i = blockIdx.x * blockDim.x + threadIdx.x;
    if (i < nnz) atomicAdd(&cnt[edges[i]], 1.0f);
}

__global__ void edge_scale_kernel(const float* __restrict__ degE,
                                  const float* __restrict__ cnt,
                                  float* scaleE, int ecnt)
{
    int i = blockIdx.x * blockDim.x + threadIdx.x;
    if (i < ecnt) scaleE[i] = degE[i] / fmaxf(cnt[i], 1.0f);
}

// V -> E scatter-add: one wave per nnz, 8 floats per lane
__global__ __launch_bounds__(256)
void scatter_ve_kernel(const float* __restrict__ X,
                       const int* __restrict__ vertex,
                       const int* __restrict__ edges,
                       float* Xe, int nnz)
{
    long t = (long)blockIdx.x * blockDim.x + threadIdx.x;
    if (t >= (long)nnz * 32) return;
    int i = (int)(t >> 5);
    int f = (int)(t & 31) << 3;
    int v = vertex[i], e = edges[i];
    const float* s = X + (size_t)v * DFEAT + f;
    float4 v0 = *(const float4*)(s);
    float4 v1 = *(const float4*)(s + 4);
    float* d = Xe + (size_t)e * DFEAT + f;
    atomicAdd(d + 0, v0.x); atomicAdd(d + 1, v0.y);
    atomicAdd(d + 2, v0.z); atomicAdd(d + 3, v0.w);
    atomicAdd(d + 4, v1.x); atomicAdd(d + 5, v1.y);
    atomicAdd(d + 6, v1.z); atomicAdd(d + 7, v1.w);
}

// E -> V scatter-add with per-edge scale folded in
__global__ __launch_bounds__(256)
void scatter_ev_kernel(const float* __restrict__ Xe,
                       const float* __restrict__ scaleE,
                       const int* __restrict__ vertex,
                       const int* __restrict__ edges,
                       float* Xv, int nnz)
{
    long t = (long)blockIdx.x * blockDim.x + threadIdx.x;
    if (t >= (long)nnz * 32) return;
    int i = (int)(t >> 5);
    int f = (int)(t & 31) << 3;
    int v = vertex[i], e = edges[i];
    float sc = scaleE[e];
    const float* s = Xe + (size_t)e * DFEAT + f;
    float4 v0 = *(const float4*)(s);
    float4 v1 = *(const float4*)(s + 4);
    float* d = Xv + (size_t)v * DFEAT + f;
    atomicAdd(d + 0, sc * v0.x); atomicAdd(d + 1, sc * v0.y);
    atomicAdd(d + 2, sc * v0.z); atomicAdd(d + 3, sc * v0.w);
    atomicAdd(d + 4, sc * v1.x); atomicAdd(d + 5, sc * v1.y);
    atomicAdd(d + 6, sc * v1.z); atomicAdd(d + 7, sc * v1.w);
}

// LayerNorm(2*degV*Xv) * g + b, then alpha-mix with X0; in-place into Xv.
// One wave32 per row, 8 floats per lane.
__global__ __launch_bounds__(256)
void ln_mix_kernel(float* __restrict__ Xv, const float* __restrict__ X0,
                   const float* __restrict__ degV,
                   const float* __restrict__ ln_g, const float* __restrict__ ln_b,
                   const float* __restrict__ alpha_arr, int layer, int nrows)
{
    int row  = (blockIdx.x * blockDim.x + threadIdx.x) >> 5;
    int lane = threadIdx.x & 31;
    if (row >= nrows) return;
    float alpha = alpha_arr[layer];
    float c = 2.0f * degV[row];
    float* xr = Xv + (size_t)row * DFEAT;
    const float* x0r = X0 + (size_t)row * DFEAT;

    const float4* p4 = (const float4*)(xr + lane * 8);
    float4 u0 = p4[0], u1 = p4[1];
    float vals[8] = { c*u0.x, c*u0.y, c*u0.z, c*u0.w, c*u1.x, c*u1.y, c*u1.z, c*u1.w };
    float s = 0.f, s2 = 0.f;
    #pragma unroll
    for (int j = 0; j < 8; ++j) { s += vals[j]; s2 += vals[j] * vals[j]; }
    #pragma unroll
    for (int m = 16; m >= 1; m >>= 1) {
        s  += __shfl_xor(s,  m, 32);
        s2 += __shfl_xor(s2, m, 32);
    }
    float mean = s * (1.0f / 256.0f);
    float var  = s2 * (1.0f / 256.0f) - mean * mean;
    float rstd = rsqrtf(var + 1e-5f);
    #pragma unroll
    for (int j = 0; j < 8; ++j) {
        int k = lane * 8 + j;
        float xn = (vals[j] - mean) * rstd * ln_g[layer * DFEAT + k] + ln_b[layer * DFEAT + k];
        xr[k] = (1.0f - alpha) * xn + alpha * x0r[k];
    }
}

// log_softmax over nclass<=40 logits (stored stride 64); one wave32 per row.
__global__ __launch_bounds__(256)
void logsoftmax_kernel(const float* __restrict__ logits, float* __restrict__ out,
                       int nrows, int nclass)
{
    int row  = (blockIdx.x * blockDim.x + threadIdx.x) >> 5;
    int lane = threadIdx.x & 31;
    if (row >= nrows) return;
    const float* L = logits + (size_t)row * 64;
    float a = (lane < nclass)      ? L[lane]      : -1e30f;
    float b = (lane + 32 < nclass) ? L[lane + 32] : -1e30f;
    float mx = fmaxf(a, b);
    #pragma unroll
    for (int m = 16; m >= 1; m >>= 1) mx = fmaxf(mx, __shfl_xor(mx, m, 32));
    float ea = (lane < nclass)      ? __expf(a - mx) : 0.f;
    float eb = (lane + 32 < nclass) ? __expf(b - mx) : 0.f;
    float s = ea + eb;
    #pragma unroll
    for (int m = 16; m >= 1; m >>= 1) s += __shfl_xor(s, m, 32);
    float lse = mx + logf(s);
    if (lane < nclass)      out[(size_t)row * nclass + lane]      = a - lse;
    if (lane + 32 < nclass) out[(size_t)row * nclass + lane + 32] = b - lse;
}

// ------------------------------------------------------------------
extern "C" void kernel_launch(void* const* d_in, const int* in_sizes, int n_in,
                              void* d_out, int out_size, void* d_ws, size_t ws_size,
                              hipStream_t stream)
{
    const float* x      = (const float*)d_in[0];
    const int*   vertex = (const int*)  d_in[1];
    const int*   edges  = (const int*)  d_in[2];
    const float* degE   = (const float*)d_in[3];
    const float* degV   = (const float*)d_in[4];
    const float* W_in   = (const float*)d_in[5];
    const float* b_in   = (const float*)d_in[6];
    const float* Wc     = (const float*)d_in[7];
    const float* alpha  = (const float*)d_in[8];
    const float* beta   = (const float*)d_in[9];
    const float* ln_g   = (const float*)d_in[10];
    const float* ln_b   = (const float*)d_in[11];
    const float* W_out  = (const float*)d_in[12];
    const float* b_out  = (const float*)d_in[13];

    const int N      = in_sizes[0] / DFEAT;
    const int NNZ    = in_sizes[1];
    const int ECNT   = in_sizes[3];
    const int NLAYER = in_sizes[8];
    const int NCLASS = in_sizes[13];

    // workspace carve-out
    char* ws = (char*)d_ws;
    size_t off = 0;
    auto carve = [&](size_t bytes) {
        void* p = ws + off;
        off = (off + bytes + 255) & ~(size_t)255;
        return p;
    };
    float* X      = (float*)carve((size_t)N    * DFEAT * 4);
    float* X0     = (float*)carve((size_t)N    * DFEAT * 4);
    float* Xe     = (float*)carve((size_t)ECNT * DFEAT * 4);
    float* Xv     = (float*)carve((size_t)N    * DFEAT * 4);
    float* cnt    = (float*)carve((size_t)ECNT * 4);
    float* scaleE = (float*)carve((size_t)ECNT * 4);
    float* logitsb= (float*)carve((size_t)N    * 64 * 4);
    (void)ws_size;

    const long nnzT    = (long)NNZ * 32;
    const int  scatBlk = (int)((nnzT + 255) / 256);
    const int  gemmBlk = (N + 63) / 64;
    const int  rowBlk  = (N + 7) / 8;   // 8 wave-rows per 256-thread block

    // edge scale = degE / max(count,1)
    zero_kernel<<<256, 256, 0, stream>>>(cnt, (long)ECNT);
    edge_count_kernel<<<(NNZ + 255) / 256, 256, 0, stream>>>(edges, cnt, NNZ);
    edge_scale_kernel<<<(ECNT + 255) / 256, 256, 0, stream>>>(degE, cnt, scaleE, ECNT);

    // X = X0 = silu(x @ W_in^T + b_in)
    gemm_kernel<DFEAT, 0><<<gemmBlk, 256, 0, stream>>>(
        x, W_in, DFEAT, b_in, nullptr, nullptr, 0, X, X0, N);

    for (int i = 0; i < NLAYER; ++i) {
        zero_kernel<<<1024, 256, 0, stream>>>(Xe, (long)ECNT * DFEAT);
        scatter_ve_kernel<<<scatBlk, 256, 0, stream>>>(X, vertex, edges, Xe, NNZ);
        zero_kernel<<<2048, 256, 0, stream>>>(Xv, (long)N * DFEAT);
        scatter_ev_kernel<<<scatBlk, 256, 0, stream>>>(Xe, scaleE, vertex, edges, Xv, NNZ);
        ln_mix_kernel<<<rowBlk, 256, 0, stream>>>(Xv, X0, degV, ln_g, ln_b, alpha, i, N);
        gemm_kernel<DFEAT, 1><<<gemmBlk, 256, 0, stream>>>(
            Xv, Wc + (size_t)i * DFEAT * DFEAT, DFEAT, nullptr, Xv, beta, i, X, nullptr, N);
    }

    // logits = X @ W_out^T + b_out (padded to 64 cols), then log_softmax
    gemm_kernel<64, 2><<<gemmBlk, 256, 0, stream>>>(
        X, W_out, NCLASS, b_out, nullptr, nullptr, 0, logitsb, nullptr, N);
    logsoftmax_kernel<<<rowBlk, 256, 0, stream>>>(logitsb, (float*)d_out, N, NCLASS);
}
